// RGAT_66649302499622
// MI455X (gfx1250) — matrix operand order repaired and checked by reading, not compiled, not run on hardware
//
#include <hip/hip_runtime.h>

// ---- CDNA5 WMMA vector types ----
typedef _Float16 v8h  __attribute__((ext_vector_type(8)));
typedef _Float16 v16h __attribute__((ext_vector_type(16)));
typedef float    v8f  __attribute__((ext_vector_type(8)));

#define N_TOTAL 100000
#define NRELS   6
#define NBASES  30
#define NEDGE   1600000
#define CH      64

static __device__ __forceinline__ v16h cat16(v8h lo, v8h hi) {
  v16h r;
#pragma unroll
  for (int i = 0; i < 8; ++i) { r[i] = lo[i]; r[8 + i] = hi[i]; }
  return r;
}

// f32 -> f16 elementwise
__global__ __launch_bounds__(256) void k_cvt16(const float* __restrict__ s,
                                               _Float16* __restrict__ d, int n) {
  int i = blockIdx.x * 256 + threadIdx.x;
  if (i < n) d[i] = (_Float16)s[i];
}

// W (K x Ncol, row-major f32) -> wT (Ncol x K, f16) so B fragments are contiguous-K loads
__global__ __launch_bounds__(256) void k_transpose_w(const float* __restrict__ W,
                                                     _Float16* __restrict__ wT,
                                                     int K, int Ncol) {
  int i = blockIdx.x * 256 + threadIdx.x;
  if (i < K * Ncol) {
    int k = i / Ncol, n = i % Ncol;
    wT[(size_t)n * K + k] = (_Float16)W[i];
  }
}

// w16T[r][o][d] = f16( sum_b att[r,b] * basis[b,d,o] )   (N-major layout for B frags)
__global__ __launch_bounds__(256) void k_wmix(const float* __restrict__ att,
                                              const float* __restrict__ basis,
                                              _Float16* __restrict__ w16T) {
  int i = blockIdx.x * 256 + threadIdx.x;
  if (i >= NRELS * CH * CH) return;
  int r = i >> 12, rem = i & 4095, o = rem >> 6, d = rem & 63;
  float s = 0.f;
#pragma unroll 5
  for (int b = 0; b < NBASES; ++b)
    s += att[r * NBASES + b] * basis[((size_t)b * CH + d) * CH + o];
  w16T[((size_t)r * CH + o) * CH + d] = (_Float16)s;
}

// Generic WMMA GEMM:  C[m,n] = act( sum_k A[m,k]*BT[n,k] + bias[n] )
// A: rows x lda (f16, row-major), BT: ncols x lda (f16, N-major).
// blockIdx.y batches (relations) via strideB/strideC. 8 waves/block, 16 rows/wave.
__global__ __launch_bounds__(256) void k_wmma_gemm(const _Float16* __restrict__ A, int lda, int rows,
                                                   const _Float16* __restrict__ BT, size_t strideB,
                                                   const float* __restrict__ bias,
                                                   float* __restrict__ C, size_t strideC,
                                                   int ldc, int ncols, int relu) {
  const _Float16* bt = BT + (size_t)blockIdx.y * strideB;
  float* cptr = C + (size_t)blockIdx.y * strideC;
  int wave = threadIdx.x >> 5;
  int lane = threadIdx.x & 31;
  int rowBase = (blockIdx.x * 8 + wave) * 16;
  if (rowBase >= rows) return;                 // wave-uniform: EXEC stays all-ones for WMMA
  int arow = rowBase + (lane & 15);
  if (arow >= rows) arow = rows - 1;           // clamp tail reads; stores are guarded
  // A 16x32 f16 fragment layout: halves 0..7 = K+{0..7}(+8 for hi lanes), 8..15 = K+16+...
  int koff  = (lane & 16) ? 8  : 0;
  // B 32x16 fragment: same K order as A (halves 0..7 = K+{0..7}(+8 for hi lanes), 8..15 = K+16+...)
  int kbb   = (lane & 16) ? 8 : 0;
  int col   = lane & 15;
  int mbase = (lane & 16) ? 8  : 0;            // D layout: VGPR v -> row 8*(lane/16)+v
  const int ntiles = ncols >> 4;
  for (int nt = 0; nt < ntiles; ++nt) {
    v8f acc;
#pragma unroll
    for (int v = 0; v < 8; ++v) acc[v] = 0.f;
    for (int ks = 0; ks < lda; ks += 32) {
      const _Float16* ap = A + (size_t)arow * lda + ks + koff;
      v16h a = cat16(*(const v8h*)ap, *(const v8h*)(ap + 16));
      const _Float16* bp = bt + (size_t)(nt * 16 + col) * lda + ks + kbb;
      v16h b = cat16(*(const v8h*)bp, *(const v8h*)(bp + 16));
      acc = __builtin_amdgcn_wmma_f32_16x16x32_f16(false, a, false, b, (short)0, acc,
                                                   false, false);
    }
    int n = nt * 16 + col;
    float bv = bias ? bias[n] : 0.f;
#pragma unroll
    for (int v = 0; v < 8; ++v) {
      int m = rowBase + mbase + v;
      if (m < rows) {
        float val = acc[v] + bv;
        if (relu) val = fmaxf(val, 0.f);
        cptr[(size_t)m * ldc + n] = val;
      }
    }
  }
}

// Row-wise LayerNorm over 64 cols; one wave per row; writes f32 (optional) + f16 copy
__global__ __launch_bounds__(256) void k_layernorm(float* __restrict__ h, _Float16* __restrict__ h16,
                                                   const float* __restrict__ g, const float* __restrict__ b,
                                                   int rows, int writeF32) {
  int wave = threadIdx.x >> 5, lane = threadIdx.x & 31;
  int row = blockIdx.x * 8 + wave;
  if (row >= rows) return;
  float* p = h + (size_t)row * CH;
  float v0 = p[lane], v1 = p[lane + 32];
  float s = v0 + v1;
#pragma unroll
  for (int m = 16; m >= 1; m >>= 1) s += __shfl_xor(s, m, 32);
  float mu = s * (1.f / CH);
  float d0 = v0 - mu, d1 = v1 - mu;
  float q = d0 * d0 + d1 * d1;
#pragma unroll
  for (int m = 16; m >= 1; m >>= 1) q += __shfl_xor(q, m, 32);
  float inv = rsqrtf(q * (1.f / CH) + 1e-5f);
  float y0 = d0 * inv * g[lane] + b[lane];
  float y1 = d1 * inv * g[lane + 32] + b[lane + 32];
  if (writeF32) { p[lane] = y0; p[lane + 32] = y1; }
  _Float16* o = h16 + (size_t)row * CH;
  o[lane] = (_Float16)y0; o[lane + 32] = (_Float16)y1;
}

// qi[r,n] = xw[r,n,:].q ; kj[r,n] = xw[r,n,:].k   (one wave per (r,n))
__global__ __launch_bounds__(256) void k_qk(const float* __restrict__ xw,
                                            const float* __restrict__ qv, const float* __restrict__ kv,
                                            float* __restrict__ qi, float* __restrict__ kj, int total) {
  int wave = threadIdx.x >> 5, lane = threadIdx.x & 31;
  int row = blockIdx.x * 8 + wave;
  if (row >= total) return;
  const float* p = xw + (size_t)row * CH;
  float a0 = p[lane], a1 = p[lane + 32];
  float sq = a0 * qv[lane] + a1 * qv[lane + 32];
  float sk = a0 * kv[lane] + a1 * kv[lane + 32];
#pragma unroll
  for (int m = 16; m >= 1; m >>= 1) { sq += __shfl_xor(sq, m, 32); sk += __shfl_xor(sk, m, 32); }
  if (lane == 0) { qi[row] = sq; kj[row] = sk; }
}

__global__ __launch_bounds__(256) void k_init(int* __restrict__ mmax, float* __restrict__ ssum,
                                              float* __restrict__ agg, int nNodes, int totalAgg) {
  int i = blockIdx.x * 256 + threadIdx.x;
  if (i < totalAgg) agg[i] = 0.f;
  if (i < nNodes) { mmax[i] = (int)0x80000000; ssum[i] = 0.f; }
}

static __device__ __forceinline__ int f2ord(float f) {
  int i = __float_as_int(f);
  return i >= 0 ? i : (i ^ 0x7fffffff);
}
static __device__ __forceinline__ float ord2f(int o) {
  return __int_as_float(o >= 0 ? o : (o ^ 0x7fffffff));
}

// alpha = leaky_relu(qi[et,dst] + kj[et,src]); segment-max via ordered-int atomicMax
__global__ __launch_bounds__(256) void k_alpha(const int* __restrict__ src, const int* __restrict__ dst,
                                               const int* __restrict__ et,
                                               const float* __restrict__ qi, const float* __restrict__ kj,
                                               float* __restrict__ alpha, int* __restrict__ mmax, int nE) {
  int e = blockIdx.x * 256 + threadIdx.x;
  if (e >= nE) return;
  int t = et[e], d = dst[e], s = src[e];
  float a = qi[(size_t)t * N_TOTAL + d] + kj[(size_t)t * N_TOTAL + s];
  a = a > 0.f ? a : 0.2f * a;
  alpha[e] = a;
  atomicMax(mmax + d, f2ord(a));
}

__global__ __launch_bounds__(256) void k_exp(const int* __restrict__ dst,
                                             const float* __restrict__ alpha, const int* __restrict__ mmax,
                                             float* __restrict__ eexp, float* __restrict__ ssum, int nE) {
  int e = blockIdx.x * 256 + threadIdx.x;
  if (e >= nE) return;
  int d = dst[e];
  float ee = __expf(alpha[e] - ord2f(mmax[d]));
  eexp[e] = ee;
  atomicAdd(ssum + d, ee);
}

// out[dst] += a * xw[et, src, :]   — 64 threads (cols) per edge, atomics land in L2
__global__ __launch_bounds__(256) void k_agg(const int* __restrict__ src, const int* __restrict__ dst,
                                             const int* __restrict__ et,
                                             const float* __restrict__ eexp, const float* __restrict__ ssum,
                                             const float* __restrict__ xw, float* __restrict__ agg, int nE) {
  int gid = blockIdx.x * 256 + threadIdx.x;
  int e = gid >> 6, c = gid & 63;
  if (e >= nE) return;
  int d = dst[e];
  float a = eexp[e] / (ssum[d] + 1e-16f);
  float msg = a * xw[((size_t)et[e] * N_TOTAL + src[e]) * CH + c];
  atomicAdd(agg + (size_t)d * CH + c, msg);
}

__global__ __launch_bounds__(256) void k_addbias(const float* __restrict__ agg, const float* __restrict__ bias,
                                                 float* __restrict__ h, _Float16* __restrict__ h16, int total) {
  int i = blockIdx.x * 256 + threadIdx.x;
  if (i >= total) return;
  float v = agg[i] + bias[i & 63];
  h[i] = v;
  h16[i] = (_Float16)v;
}

extern "C" void kernel_launch(void* const* d_in, const int* in_sizes, int n_in,
                              void* d_out, int out_size, void* d_ws, size_t ws_size,
                              hipStream_t stream) {
  (void)in_sizes; (void)n_in; (void)out_size; (void)ws_size;
  const float* x0    = (const float*)d_in[0];
  const float* l0W   = (const float*)d_in[1];
  const float* l0b   = (const float*)d_in[2];
  const float* l0g   = (const float*)d_in[3];
  const float* l0be  = (const float*)d_in[4];
  const float* x1    = (const float*)d_in[5];
  const float* l1W   = (const float*)d_in[6];
  const float* l1b   = (const float*)d_in[7];
  const float* l1g   = (const float*)d_in[8];
  const float* l1be  = (const float*)d_in[9];
  const float* x2    = (const float*)d_in[10];
  const float* l2W   = (const float*)d_in[11];
  const float* l2b   = (const float*)d_in[12];
  const float* l2g   = (const float*)d_in[13];
  const float* l2be  = (const float*)d_in[14];
  const int*   eidx  = (const int*)d_in[15];
  const int*   etyp  = (const int*)d_in[16];
  const float* att[2]   = {(const float*)d_in[17], (const float*)d_in[22]};
  const float* basis[2] = {(const float*)d_in[18], (const float*)d_in[23]};
  const float* qv[2]    = {(const float*)d_in[19], (const float*)d_in[24]};
  const float* kv[2]    = {(const float*)d_in[20], (const float*)d_in[25]};
  const float* cbias[2] = {(const float*)d_in[21], (const float*)d_in[26]};
  const float* lng   = (const float*)d_in[27];
  const float* lnb   = (const float*)d_in[28];
  const float* outW  = (const float*)d_in[29];
  const float* outb  = (const float*)d_in[30];

  const int* esrc = eidx;
  const int* edst = eidx + NEDGE;

  // ---- workspace carve-out (~250 MB) ----
  char* wsp = (char*)d_ws;
  auto alloc = [&](size_t bytes) -> void* {
    void* r = (void*)wsp; wsp += (bytes + 255) & ~(size_t)255; return r;
  };
  _Float16* x16_0 = (_Float16*)alloc((size_t)40000 * 128 * 2);
  _Float16* x16_1 = (_Float16*)alloc((size_t)35000 * 64 * 2);
  _Float16* x16_2 = (_Float16*)alloc((size_t)25000 * 256 * 2);
  _Float16* wT0   = (_Float16*)alloc((size_t)64 * 128 * 2);
  _Float16* wT1   = (_Float16*)alloc((size_t)64 * 64 * 2);
  _Float16* wT2   = (_Float16*)alloc((size_t)64 * 256 * 2);
  _Float16* wTout = (_Float16*)alloc((size_t)32 * 64 * 2);
  float*    h     = (float*)   alloc((size_t)N_TOTAL * CH * 4);
  _Float16* h16   = (_Float16*)alloc((size_t)N_TOTAL * CH * 2);
  _Float16* w16T  = (_Float16*)alloc((size_t)NRELS * CH * CH * 2);
  float*    xw    = (float*)   alloc((size_t)NRELS * N_TOTAL * CH * 4);
  float*    qi    = (float*)   alloc((size_t)NRELS * N_TOTAL * 4);
  float*    kj    = (float*)   alloc((size_t)NRELS * N_TOTAL * 4);
  float*    alp   = (float*)   alloc((size_t)NEDGE * 4);
  float*    eexp  = (float*)   alloc((size_t)NEDGE * 4);
  int*      mmax  = (int*)     alloc((size_t)N_TOTAL * 4);
  float*    ssum  = (float*)   alloc((size_t)N_TOTAL * 4);
  float*    agg   = (float*)   alloc((size_t)N_TOTAL * CH * 4);

  auto cdiv = [](int a, int b) { return (a + b - 1) / b; };

  // ---- input conversions ----
  k_cvt16<<<cdiv(40000 * 128, 256), 256, 0, stream>>>(x0, x16_0, 40000 * 128);
  k_cvt16<<<cdiv(35000 * 64, 256), 256, 0, stream>>>(x1, x16_1, 35000 * 64);
  k_cvt16<<<cdiv(25000 * 256, 256), 256, 0, stream>>>(x2, x16_2, 25000 * 256);
  k_transpose_w<<<cdiv(128 * 64, 256), 256, 0, stream>>>(l0W, wT0, 128, 64);
  k_transpose_w<<<cdiv(64 * 64, 256), 256, 0, stream>>>(l1W, wT1, 64, 64);
  k_transpose_w<<<cdiv(256 * 64, 256), 256, 0, stream>>>(l2W, wT2, 256, 64);
  k_transpose_w<<<cdiv(64 * 32, 256), 256, 0, stream>>>(outW, wTout, 64, 32);

  // ---- per-type projection: GEMM + bias + ReLU, then LayerNorm -> h/h16 ----
  k_wmma_gemm<<<cdiv(cdiv(40000, 16), 8), 256, 0, stream>>>(x16_0, 128, 40000, wT0, 0, l0b,
                                                            h, 0, 64, 64, 1);
  k_wmma_gemm<<<cdiv(cdiv(35000, 16), 8), 256, 0, stream>>>(x16_1, 64, 35000, wT1, 0, l1b,
                                                            h + (size_t)40000 * 64, 0, 64, 64, 1);
  k_wmma_gemm<<<cdiv(cdiv(25000, 16), 8), 256, 0, stream>>>(x16_2, 256, 25000, wT2, 0, l2b,
                                                            h + (size_t)75000 * 64, 0, 64, 64, 1);
  k_layernorm<<<cdiv(40000, 8), 256, 0, stream>>>(h, h16, l0g, l0be, 40000, 1);
  k_layernorm<<<cdiv(35000, 8), 256, 0, stream>>>(h + (size_t)40000 * 64, h16 + (size_t)40000 * 64,
                                                  l1g, l1be, 35000, 1);
  k_layernorm<<<cdiv(25000, 8), 256, 0, stream>>>(h + (size_t)75000 * 64, h16 + (size_t)75000 * 64,
                                                  l2g, l2be, 25000, 1);

  // ---- two RGAT convolutions ----
  for (int cv = 0; cv < 2; ++cv) {
    k_wmix<<<cdiv(NRELS * CH * CH, 256), 256, 0, stream>>>(att[cv], basis[cv], w16T);
    k_init<<<cdiv(N_TOTAL * CH, 256), 256, 0, stream>>>(mmax, ssum, agg, N_TOTAL, N_TOTAL * CH);
    dim3 gxw(cdiv(cdiv(N_TOTAL, 16), 8), NRELS);
    k_wmma_gemm<<<gxw, 256, 0, stream>>>(h16, 64, N_TOTAL, w16T, (size_t)CH * CH, nullptr,
                                         xw, (size_t)N_TOTAL * CH, 64, 64, 0);
    k_qk<<<cdiv(NRELS * N_TOTAL, 8), 256, 0, stream>>>(xw, qv[cv], kv[cv], qi, kj, NRELS * N_TOTAL);
    k_alpha<<<cdiv(NEDGE, 256), 256, 0, stream>>>(esrc, edst, etyp, qi, kj, alp, mmax, NEDGE);
    k_exp<<<cdiv(NEDGE, 256), 256, 0, stream>>>(edst, alp, mmax, eexp, ssum, NEDGE);
    k_agg<<<cdiv(NEDGE * 64, 256), 256, 0, stream>>>(esrc, edst, etyp, eexp, ssum, xw, agg, NEDGE);
    k_addbias<<<cdiv(N_TOTAL * CH, 256), 256, 0, stream>>>(agg, cbias[cv], h, h16, N_TOTAL * CH);
  }

  // ---- final LayerNorm (first 40000 rows) + output GEMM (64 -> 32) ----
  k_layernorm<<<cdiv(40000, 8), 256, 0, stream>>>(h, h16, lng, lnb, 40000, 0);
  k_wmma_gemm<<<cdiv(cdiv(40000, 16), 8), 256, 0, stream>>>(h16, 64, 40000, wTout, 0, outb,
                                                            (float*)d_out, 0, 32, 32, 0);
}
